// GraphGatedGCNModel_88287347737110
// MI455X (gfx1250) — matrix operand order repaired and checked by reading, not compiled, not run on hardware
//
#include <hip/hip_runtime.h>

#define NN 100000
#define EE 800000
#define HH 32
#define LL 4

typedef __attribute__((ext_vector_type(16))) __bf16 v16bf;
typedef __attribute__((ext_vector_type(8)))  float  v8f;
typedef __attribute__((ext_vector_type(4)))  float  v4f;

// CDNA5 async global->LDS copies (ASYNCcnt-tracked), guarded so the file
// compiles on toolchains that lack the builtins.
#if defined(__has_builtin)
#if __has_builtin(__builtin_amdgcn_global_load_async_to_lds_b128) && \
    __has_builtin(__builtin_amdgcn_s_wait_asynccnt)
#define HAVE_ASYNC_LDS 1
#endif
#endif
#ifndef HAVE_ASYNC_LDS
#define HAVE_ASYNC_LDS 0
#endif

// builtin signature (from hipcc diagnostic): (v4i AS1*, v4i AS3*, imm int, imm int)
typedef int v4i_t __attribute__((vector_size(16)));
typedef __attribute__((address_space(1))) v4i_t* g4_t;
typedef __attribute__((address_space(3))) v4i_t* l4_t;

__device__ __forceinline__ void async_copy16(const float* gsrc, float* ldst) {
#if HAVE_ASYNC_LDS
    // generic->AS pointer via integer round trip (flat LDS addr low bits == LDS
    // offset; flat global addr == global addr). Int->pointer casts are always
    // accepted and give exactly the target pointer type.
    __builtin_amdgcn_global_load_async_to_lds_b128(
        (g4_t)(unsigned long long)(uintptr_t)gsrc,
        (l4_t)(unsigned int)(uintptr_t)ldst, 0, 0);
#else
    *(v4f*)ldst = *(const v4f*)gsrc;
#endif
}

__device__ __forceinline__ void async_wait0() {
#if HAVE_ASYNC_LDS
    __builtin_amdgcn_s_wait_asynccnt(0);
#endif
}

// ---------------------------------------------------------------------------
// zero scratch region
__global__ void __launch_bounds__(256) zero_kernel(float* __restrict__ p, int n) {
    int t = blockIdx.x * 256 + threadIdx.x;
    if (t < n) p[t] = 0.f;
}

// ---------------------------------------------------------------------------
// input projection: out[i,j] = v[i] * w[j] + b[j]   (x is [rows,1])
__global__ void __launch_bounds__(256) proj_kernel(const float* __restrict__ v,
                                                   const float* __restrict__ w,
                                                   const float* __restrict__ b,
                                                   float* __restrict__ out, int total) {
    int t = blockIdx.x * 256 + threadIdx.x;
    if (t >= total) return;
    int j = t & 31;
    out[t] = v[t >> 5] * w[j] + b[j];
}

// ---------------------------------------------------------------------------
// X[rows,32] @ [W0|W1|..] (each 32x32 row-major) + bias -> out[rows, NBLK*32]
// bf16 WMMA 16x16x32, f32 accumulate. One WMMA per 16x16 output tile (K=32).
template <int NBLK>
__global__ void __launch_bounds__(256) gemm32_wmma(
    const float* __restrict__ X,
    const float* __restrict__ w0, const float* __restrict__ w1,
    const float* __restrict__ w2, const float* __restrict__ w3,
    const float* __restrict__ b0, const float* __restrict__ b1,
    const float* __restrict__ b2, const float* __restrict__ b3,
    float* __restrict__ out, int rows)
{
    const int ncols = NBLK * 32;
    __shared__ __bf16 wl[32 * NBLK * 32];
    __shared__ float  bl[NBLK * 32];
    const float* wsrc[4] = {w0, w1, w2, w3};
    const float* bsrc[4] = {b0, b1, b2, b3};
    for (int idx = threadIdx.x; idx < 32 * ncols; idx += 256) {
        int k = idx / ncols, j = idx - k * ncols;
        wl[idx] = (__bf16)wsrc[j >> 5][k * 32 + (j & 31)];
    }
    for (int j = threadIdx.x; j < ncols; j += 256)
        bl[j] = bsrc[j >> 5][j & 31];
    __syncthreads();

    const int lane = threadIdx.x & 31;
    const int n16  = lane & 15;

    // B fragments: lanes 0-15 hold K=0..15, lanes 16-31 hold K=16..31 of column n16
    v16bf bf[NBLK * 2];
    {
        const int kb = (lane < 16) ? 0 : 16;
        for (int t = 0; t < NBLK * 2; ++t) {
            v16bf b;
            const int col = t * 16 + n16;
            for (int u = 0; u < 16; ++u) b[u] = wl[(kb + u) * ncols + col];
            bf[t] = b;
        }
    }

    const int waveId = blockIdx.x * 8 + (threadIdx.x >> 5);
    const int nwaves = gridDim.x * 8;
    const int nslab  = rows >> 4;
    const int ka     = (lane < 16) ? 0 : 16;  // A layout K base
    const int rb     = (lane < 16) ? 0 : 8;   // D layout row base

    for (int s = waveId; s < nslab; s += nwaves) {
        const float* xr = X + (size_t)(s * 16 + n16) * 32;
        v16bf a;
        for (int u = 0; u < 8; ++u) {
            a[u]     = (__bf16)xr[ka + u];        // K = ka .. ka+7
            a[u + 8] = (__bf16)xr[ka + 8 + u];    // K = ka+8 .. ka+15
        }
        for (int t = 0; t < NBLK * 2; ++t) {
            v8f c;
            const float bias = bl[t * 16 + n16];
            for (int i = 0; i < 8; ++i) c[i] = bias;
            c = __builtin_amdgcn_wmma_f32_16x16x32_bf16(false, a, false, bf[t],
                                                        (short)0, c, false, false);
            float* orow = out + (size_t)(s * 16 + rb) * ncols + t * 16 + n16;
            for (int i = 0; i < 8; ++i) orow[(size_t)i * ncols] = c[i];
        }
    }
}

// ---------------------------------------------------------------------------
// edge update: e_new = Ce + Ah[dst] + Bh[src]; sigma=sigmoid(e_new);
// atomic scatter num/den; fused BN stats for e_new into red[64..127]
__global__ void __launch_bounds__(256) edge_update_kernel(
    const float* __restrict__ nodeG, const int* __restrict__ ei,
    float* __restrict__ enew, float* __restrict__ num, float* __restrict__ den,
    float* __restrict__ red)
{
    const int t = blockIdx.x * 256 + threadIdx.x;
    const int j = threadIdx.x & 31;
    float v = 0.f, vsq = 0.f;
    if (t < EE * 32) {
        const int i = t >> 5;
        const int s = ei[i];
        const int d = ei[EE + i];
        const float val = enew[t] + nodeG[(size_t)d * 128 + j]        // Ah[dst]
                                  + nodeG[(size_t)s * 128 + 32 + j];  // Bh[src]
        enew[t] = val;
        const float sg = 1.f / (1.f + __expf(-val));
        atomicAdd(&num[(size_t)d * 32 + j], sg * nodeG[(size_t)s * 128 + 96 + j]); // Vh[src]
        atomicAdd(&den[(size_t)d * 32 + j], sg);
        v = val; vsq = val * val;
    }
    __shared__ float rs[256], rq[256];
    rs[threadIdx.x] = v; rq[threadIdx.x] = vsq;
    __syncthreads();
    if (threadIdx.x < 32) {
        float a = 0.f, b = 0.f;
        for (int r = 0; r < 8; ++r) { a += rs[r * 32 + threadIdx.x]; b += rq[r * 32 + threadIdx.x]; }
        atomicAdd(&red[64 + threadIdx.x], a);
        atomicAdd(&red[96 + threadIdx.x], b);
    }
}

// ---------------------------------------------------------------------------
// node update: h_new = Uh + num/(den+eps) (written in-place over num);
// fused BN stats for h_new into red[0..63]
__global__ void __launch_bounds__(256) node_update_kernel(
    const float* __restrict__ nodeG, float* __restrict__ num,
    const float* __restrict__ den, float* __restrict__ red)
{
    const int t = blockIdx.x * 256 + threadIdx.x;
    const int j = threadIdx.x & 31;
    float v = 0.f, vsq = 0.f;
    if (t < NN * 32) {
        const int n = t >> 5;
        const float val = nodeG[(size_t)n * 128 + 64 + j] + num[t] / (den[t] + 1e-6f);
        num[t] = val;   // reuse num as h_new
        v = val; vsq = val * val;
    }
    __shared__ float rs[256], rq[256];
    rs[threadIdx.x] = v; rq[threadIdx.x] = vsq;
    __syncthreads();
    if (threadIdx.x < 32) {
        float a = 0.f, b = 0.f;
        for (int r = 0; r < 8; ++r) { a += rs[r * 32 + threadIdx.x]; b += rq[r * 32 + threadIdx.x]; }
        atomicAdd(&red[threadIdx.x], a);
        atomicAdd(&red[32 + threadIdx.x], b);
    }
}

// ---------------------------------------------------------------------------
// x += relu( (xnew - mu) * rsqrt(var+eps) * g + b )
__global__ void __launch_bounds__(256) bn_apply_kernel(
    float* __restrict__ xio, const float* __restrict__ xnew,
    const float* __restrict__ red, int roff,
    const float* __restrict__ g, const float* __restrict__ b,
    float invcnt, int total)
{
    const int t = blockIdx.x * 256 + threadIdx.x;
    if (t >= total) return;
    const int j = t & 31;
    const float mu  = red[roff + j] * invcnt;
    const float var = red[roff + 32 + j] * invcnt - mu * mu;
    const float v = (xnew[t] - mu) * rsqrtf(var + 1e-5f) * g[j] + b[j];
    xio[t] += v > 0.f ? v : 0.f;
}

// ---------------------------------------------------------------------------
// scorer: z=[h_src|h_dst|e] (E x 96) -> relu(z@W1+b1) @ W2 + b2
// Per block: 128 edges staged into LDS with ASYNC global->LDS B128 copies
// (ASYNCcnt), then each wave runs a 16-edge slab with 6 chained WMMAs
// (3 K-chunks x 2 col tiles), then a 32-wide dot with W2.
__global__ void __launch_bounds__(256) score_kernel(
    const float* __restrict__ h, const float* __restrict__ efeat,
    const int* __restrict__ ei,
    const float* __restrict__ W1, const float* __restrict__ b1,
    const float* __restrict__ W2, const float* __restrict__ b2,
    float* __restrict__ out)
{
    __shared__ float zt[128][100];   // 128 edges x 96 (row = 400 B, 16B-aligned)
    __shared__ float sc[128][33];    // relu(hidden)
    __shared__ float w1l[96 * 32];
    __shared__ float w2l[32];
    for (int idx = threadIdx.x; idx < 96 * 32; idx += 256) w1l[idx] = W1[idx];
    if (threadIdx.x < 32) w2l[threadIdx.x] = W2[threadIdx.x];
    __syncthreads();

    const int lane = threadIdx.x & 31;
    const int wave = threadIdx.x >> 5;
    const int n16  = lane & 15;

    v16bf bf[6];
    {
        const int kb = (lane < 16) ? 0 : 16;
        for (int c = 0; c < 3; ++c)
            for (int t2 = 0; t2 < 2; ++t2) {
                v16bf b;
                for (int u = 0; u < 16; ++u)
                    b[u] = (__bf16)w1l[(c * 32 + kb + u) * 32 + t2 * 16 + n16];
                bf[c * 2 + t2] = b;
            }
    }
    const float bias0 = b1[n16], bias1 = b1[16 + n16];
    const float bb = b2[0];
    const int ka = (lane < 16) ? 0 : 16;
    const int ntiles = EE / 128;

    for (int tile = blockIdx.x; tile < ntiles; tile += gridDim.x) {
        const int ebase = tile * 128;
        __syncthreads();   // previous iteration's zt/sc readers are done

        // ---- async gather stage: 128 edges x 3 rows x 8 float4 chunks ----
        for (int c = threadIdx.x; c < 128 * 24; c += 256) {
            const int edge = c / 24;
            const int w    = c - edge * 24;
            const int seg  = w >> 3;            // 0: h[src], 1: h[dst], 2: e
            const int q    = w & 7;             // float4 index within the row
            const int eid  = ebase + edge;
            const float* gsrc;
            if (seg == 0)      gsrc = h     + (size_t)ei[eid] * 32      + q * 4;
            else if (seg == 1) gsrc = h     + (size_t)ei[EE + eid] * 32 + q * 4;
            else               gsrc = efeat + (size_t)eid * 32          + q * 4;
            async_copy16(gsrc, &zt[edge][seg * 32 + q * 4]);
        }
        async_wait0();
        __syncthreads();

        const int rr = wave * 16 + n16;
        v8f acc0, acc1;
        for (int i = 0; i < 8; ++i) { acc0[i] = bias0; acc1[i] = bias1; }
        for (int c = 0; c < 3; ++c) {
            v16bf a;
            for (int u = 0; u < 8; ++u) {
                a[u]     = (__bf16)zt[rr][c * 32 + ka + u];
                a[u + 8] = (__bf16)zt[rr][c * 32 + ka + 8 + u];
            }
            acc0 = __builtin_amdgcn_wmma_f32_16x16x32_bf16(false, a, false, bf[c * 2],
                                                           (short)0, acc0, false, false);
            acc1 = __builtin_amdgcn_wmma_f32_16x16x32_bf16(false, a, false, bf[c * 2 + 1],
                                                           (short)0, acc1, false, false);
        }
        const int rb = wave * 16 + ((lane < 16) ? 0 : 8);
        for (int i = 0; i < 8; ++i) {
            const float v0 = acc0[i], v1 = acc1[i];
            sc[rb + i][n16]      = v0 > 0.f ? v0 : 0.f;
            sc[rb + i][16 + n16] = v1 > 0.f ? v1 : 0.f;
        }
        __syncthreads();
        if (threadIdx.x < 128) {
            float accs = bb;
            for (int c2 = 0; c2 < 32; ++c2) accs += sc[threadIdx.x][c2] * w2l[c2];
            out[ebase + threadIdx.x] = accs;
        }
    }
}

// ---------------------------------------------------------------------------
extern "C" void kernel_launch(void* const* d_in, const int* in_sizes, int n_in,
                              void* d_out, int out_size, void* d_ws, size_t ws_size,
                              hipStream_t stream)
{
    (void)in_sizes; (void)n_in; (void)out_size; (void)ws_size;
    const float* x    = (const float*)d_in[0];
    const float* ein  = (const float*)d_in[1];
    const int*   eidx = (const int*)d_in[2];
    const float* pe_w = (const float*)d_in[3];
    const float* pe_b = (const float*)d_in[4];
    const float* ed_w = (const float*)d_in[5];
    const float* ed_b = (const float*)d_in[6];
    const float* A_w = (const float*)d_in[7];  const float* A_b = (const float*)d_in[8];
    const float* B_w = (const float*)d_in[9];  const float* B_b = (const float*)d_in[10];
    const float* C_w = (const float*)d_in[11]; const float* C_b = (const float*)d_in[12];
    const float* U_w = (const float*)d_in[13]; const float* U_b = (const float*)d_in[14];
    const float* V_w = (const float*)d_in[15]; const float* V_b = (const float*)d_in[16];
    const float* bn_h_g = (const float*)d_in[17]; const float* bn_h_b = (const float*)d_in[18];
    const float* bn_e_g = (const float*)d_in[19]; const float* bn_e_b = (const float*)d_in[20];
    const float* W1_w = (const float*)d_in[21]; const float* W1_b = (const float*)d_in[22];
    const float* W2_w = (const float*)d_in[23]; const float* W2_b = (const float*)d_in[24];
    float* out = (float*)d_out;

    float* ws = (float*)d_ws;
    size_t o = 0;
    float* h     = ws + o; o += (size_t)NN * 32;    // node features
    float* efeat = ws + o; o += (size_t)EE * 32;    // edge features
    float* nodeG = ws + o; o += (size_t)NN * 128;   // [Ah|Bh|Uh|Vh]
    float* enew  = ws + o; o += (size_t)EE * 32;    // Ce then e_new
    float* num   = ws + o; o += (size_t)NN * 32;    // scatter num, then h_new
    float* den   = ws + o; o += (size_t)NN * 32;    // scatter den
    float* red   = ws + o; o += 128;                // BN stats (h sum/sq, e sum/sq)

    proj_kernel<<<(NN * 32) / 256, 256, 0, stream>>>(x, pe_w, pe_b, h, NN * 32);
    proj_kernel<<<(EE * 32) / 256, 256, 0, stream>>>(ein, ed_w, ed_b, efeat, EE * 32);

    for (int l = 0; l < LL; ++l) {
        const int zn = NN * 64 + 128;   // num, den, red are contiguous
        zero_kernel<<<(zn + 255) / 256, 256, 0, stream>>>(num, zn);

        gemm32_wmma<4><<<782, 256, 0, stream>>>(h,
            A_w + l * 1024, B_w + l * 1024, U_w + l * 1024, V_w + l * 1024,
            A_b + l * 32,   B_b + l * 32,   U_b + l * 32,   V_b + l * 32,
            nodeG, NN);
        gemm32_wmma<1><<<6250, 256, 0, stream>>>(efeat,
            C_w + l * 1024, C_w + l * 1024, C_w + l * 1024, C_w + l * 1024,
            C_b + l * 32,   C_b + l * 32,   C_b + l * 32,   C_b + l * 32,
            enew, EE);

        edge_update_kernel<<<(EE * 32) / 256, 256, 0, stream>>>(nodeG, eidx, enew, num, den, red);
        node_update_kernel<<<(NN * 32) / 256, 256, 0, stream>>>(nodeG, num, den, red);

        bn_apply_kernel<<<(NN * 32) / 256, 256, 0, stream>>>(h, num, red, 0,
            bn_h_g + l * 32, bn_h_b + l * 32, 1.0f / NN, NN * 32);
        bn_apply_kernel<<<(EE * 32) / 256, 256, 0, stream>>>(efeat, enew, red, 64,
            bn_e_g + l * 32, bn_e_b + l * 32, 1.0f / EE, EE * 32);
    }

    score_kernel<<<6250, 256, 0, stream>>>(h, efeat, eidx, W1_w, W1_b, W2_w, W2_b, out);
}
